// GCODERNN_32933809225745
// MI455X (gfx1250) — matrix operand order repaired and checked by reading, not compiled, not run on hardware
//
#include <hip/hip_runtime.h>
#include <hip/hip_bf16.h>
#include <math.h>

// ---------------------------------------------------------------------------
// GC-ODE-RNN for MI455X (gfx1250, wave32, WMMA).
// All matmuls run through v_wmma_f32_16x16x32_f16 (f16 in, f32 accumulate).
// A (8x1024x1024) is converted to f16 once (16MB -> resident in 192MB L2).
// Hidden states are kept BOTH row-major f16 [N][D] (WMMA A-operand) and
// transposed f16 [D][N] (WMMA B-operand) so every fragment load is b128s.
// ---------------------------------------------------------------------------

#define NB   8      // batch
#define NT   32     // time steps
#define NN   1024   // nodes
#define DIN  64     // input feature dim
#define DR   128    // rnn / ode hidden dim
#define DXH  192    // DIN + DR

typedef __attribute__((ext_vector_type(16))) _Float16 v16h;
typedef __attribute__((ext_vector_type(8)))  float    v8f;

// ---------------------------------------------------------------------------
// k1_spmm: G[b] = A[b] @ H[b]      (the N^2-dominant graph aggregation)
//   A16  : f16 [B][NN][NN] row-major
//   B-operand source = H^T rows: row d comes from src0 (d<split) or src1.
//   Output G f16 row-major [NN][ldc] per batch.
// One wave computes a 16x64 output tile: 1 A-fragment feeds 4 WMMAs.
// ---------------------------------------------------------------------------
__global__ __launch_bounds__(256) void k1_spmm(
    const _Float16* __restrict__ A16,
    const _Float16* __restrict__ src0, long s0_bstr,
    const _Float16* __restrict__ src1, long s1_bstr,
    int split,
    _Float16* __restrict__ G, int ldc, long g_bstr)
{
    const int lane = threadIdx.x & 31;
    const int wave = threadIdx.x >> 5;          // 0..7
    const int b    = blockIdx.z;
    const int m0   = (blockIdx.x * 8 + wave) * 16;
    const int n0   = blockIdx.y * 64;
    const int r    = lane & 15;
    const int half = lane >> 4;

    // A operand: 16x32 f16 tile. lane<16: row m0+r, K = k0+[0..7] & k0+16+[0..7]
    //            lane>=16: same rows, K shifted by 8 (per ISA A layout).
    const _Float16* arow = A16 + (long)b * NN * NN + (long)(m0 + r) * NN + half * 8;

    // B operand rows: per lane a fixed H^T row (feature d), K contiguous.
    const _Float16* brow[4];
#pragma unroll
    for (int g = 0; g < 4; ++g) {
        int d = n0 + g * 16 + r;
        brow[g] = (d < split)
            ? src0 + (long)b * s0_bstr + (long)d * NN + half * 8
            : src1 + (long)b * s1_bstr + (long)(d - split) * NN + half * 8;
    }

    v8f acc[4] = {};
    for (int k0 = 0; k0 < NN; k0 += 32) {
        v16h a;
        ((float4*)&a)[0] = *(const float4*)(arow + k0);        // K = k0+half*8 .. +7
        ((float4*)&a)[1] = *(const float4*)(arow + k0 + 16);   // K = +16
#pragma unroll
        for (int g = 0; g < 4; ++g) {
            v16h bf;
            ((float4*)&bf)[0] = *(const float4*)(brow[g] + k0);      // K = k0+half*8 .. +7
            ((float4*)&bf)[1] = *(const float4*)(brow[g] + k0 + 16);
            acc[g] = __builtin_amdgcn_wmma_f32_16x16x32_f16(
                false, a, false, bf, (short)0, acc[g], false, false);
        }
    }

    _Float16* Gb = G + (long)b * g_bstr;
#pragma unroll
    for (int g = 0; g < 4; ++g) {
        int n = n0 + g * 16 + r;
#pragma unroll
        for (int i = 0; i < 8; ++i) {
            int m = m0 + half * 8 + i;          // C layout: VGPR i -> M = 8*half + i
            Gb[(long)m * ldc + n] = (_Float16)acc[g][i];
        }
    }
}

// ---------------------------------------------------------------------------
// k2_dense: out = act( G @ W + bias )   (the small per-node weight multiply)
//   A-operand rows from s0 (k<kl0) or s1 (fused c-gate: [A@x | A@rh] @ Wh).
//   B-operand = W^T f16 [Dout][Din] (K contiguous per output column).
//   ACT: 0 none, 1 tanh, 2 sigmoid.  OUTM: 0 = f32 row-major, 1 = f16 [Dout][NN].
// ---------------------------------------------------------------------------
template<int ACT, int OUTM>
__global__ __launch_bounds__(256) void k2_dense(
    const _Float16* __restrict__ s0, int ld0, long s0_bstr, int kl0,
    const _Float16* __restrict__ s1, int ld1, long s1_bstr,
    const _Float16* __restrict__ WT, int Din,
    const float* __restrict__ bias,
    float* __restrict__ outF, int ldo, long o_bstr,
    _Float16* __restrict__ outT, long t_bstr)
{
    const int lane = threadIdx.x & 31;
    const int wave = threadIdx.x >> 5;
    const int b    = blockIdx.z;
    const int m0   = (blockIdx.x * 8 + wave) * 16;
    const int n0   = blockIdx.y * 16;
    const int r    = lane & 15;
    const int half = lane >> 4;

    const _Float16* brow = WT + (long)(n0 + r) * Din + half * 8;

    v8f acc = {};
    for (int k0 = 0; k0 < Din; k0 += 32) {
        const _Float16* ap;
        if (k0 < kl0)
            ap = s0 + (long)b * s0_bstr + (long)(m0 + r) * ld0 + k0;
        else
            ap = s1 + (long)b * s1_bstr + (long)(m0 + r) * ld1 + (k0 - kl0);
        v16h a;
        ((float4*)&a)[0] = *(const float4*)(ap + half * 8);
        ((float4*)&a)[1] = *(const float4*)(ap + half * 8 + 16);
        v16h bf;
        ((float4*)&bf)[0] = *(const float4*)(brow + k0);
        ((float4*)&bf)[1] = *(const float4*)(brow + k0 + 16);
        acc = __builtin_amdgcn_wmma_f32_16x16x32_f16(
            false, a, false, bf, (short)0, acc, false, false);
    }

    const int  nout = n0 + r;
    const float bv  = bias[nout];
#pragma unroll
    for (int i = 0; i < 8; ++i) {
        float v = acc[i] + bv;
        if (ACT == 1)      v = tanhf(v);
        else if (ACT == 2) v = 1.0f / (1.0f + __expf(-v));
        int m = m0 + half * 8 + i;
        if (OUTM == 0) outF[(long)b * o_bstr + (long)m * ldo + nout] = v;
        else           outT[(long)b * t_bstr + (long)nout * NN + m]  = (_Float16)v;
    }
}

// ----------------------------- prep kernels --------------------------------
__global__ void cvt_a_kernel(const float* __restrict__ A, _Float16* __restrict__ A16, long total) {
    long i = (long)blockIdx.x * blockDim.x + threadIdx.x;
    if (i < total) A16[i] = (_Float16)A[i];
}

__global__ void wtr_kernel(const float* __restrict__ W, _Float16* __restrict__ WT, int Din, int Dout) {
    int i = blockIdx.x * blockDim.x + threadIdx.x;
    if (i >= Din * Dout) return;
    int din = i / Dout, dout = i % Dout;
    WT[dout * Din + din] = (_Float16)W[i];                 // WT[Dout][Din]
}

__global__ void init_h_kernel(const float* __restrict__ h0,
                              float* __restrict__ hv, _Float16* __restrict__ hvT) {
    long i = (long)blockIdx.x * blockDim.x + threadIdx.x;
    if (i >= (long)NB * NN * DR) return;
    int d = i & (DR - 1);
    long nn = i >> 7; int n = nn & (NN - 1); int b = (int)(nn >> 10);
    float v = h0[d];
    hv[i] = v;
    hvT[((long)b * DR + d) * NN + n] = (_Float16)v;
}

__global__ void obs_kernel(const float* __restrict__ masks, float* __restrict__ obs) {
    __shared__ float s[256];
    const float* p = masks + (long)blockIdx.x * NN * DIN;
    float sum = 0.f;
    for (int i = threadIdx.x; i < NN * DIN; i += 256) sum += fabsf(p[i]);
    s[threadIdx.x] = sum; __syncthreads();
    for (int o = 128; o > 0; o >>= 1) {
        if ((int)threadIdx.x < o) s[threadIdx.x] += s[threadIdx.x + o];
        __syncthreads();
    }
    if (threadIdx.x == 0) obs[blockIdx.x] = (s[0] > 1e-4f) ? 1.f : 0.f;
}

// XTt[b][d][n] = (f16)(values[b][t][n][d] * masks[b][t][n][d])
__global__ void xt_kernel(const float* __restrict__ vals, const float* __restrict__ masks,
                          _Float16* __restrict__ XTt, int t) {
    long i = (long)blockIdx.x * blockDim.x + threadIdx.x;
    if (i >= (long)NB * DIN * NN) return;
    int n = (int)(i % NN); long rest = i / NN;
    int d = (int)(rest % DIN); int b = (int)(rest / DIN);
    long src = (((long)b * NT + t) * NN + n) * DIN + d;
    XTt[i] = (_Float16)(vals[src] * masks[src]);
}

// --------------------------- elementwise kernels ---------------------------
// hv1 = hv + DT*ode ; write f32, f16 row-major, f16 transposed
__global__ void ew_ode_kernel(const float* __restrict__ hv, const float* __restrict__ ode,
                              float* __restrict__ o32, _Float16* __restrict__ o16rm,
                              _Float16* __restrict__ o16t) {
    long i = (long)blockIdx.x * blockDim.x + threadIdx.x;
    if (i >= (long)NB * NN * DR) return;
    float v = hv[i] + 0.25f * ode[i];
    o32[i] = v;
    if (o16rm) o16rm[i] = (_Float16)v;
    int d = i & (DR - 1); long nn = i >> 7; int n = nn & (NN - 1); int b = (int)(nn >> 10);
    o16t[((long)b * DR + d) * NN + n] = (_Float16)v;
}

// rhT[b][d][n] = (f16)(r[b][n][d] * hv1[b][n][d])
__global__ void ew_rh_kernel(const float* __restrict__ rg, const float* __restrict__ hv1,
                             _Float16* __restrict__ rhT) {
    long i = (long)blockIdx.x * blockDim.x + threadIdx.x;
    if (i >= (long)NB * NN * DR) return;
    int d = i & (DR - 1); long nn = i >> 7; int n = nn & (NN - 1); int b = (int)(nn >> 10);
    rhT[((long)b * DR + d) * NN + n] = (_Float16)(rg[i] * hv1[i]);
}

// hv = obs ? ((1-z)*hv1 + z*c) : hv1 ; write f32 + f16 transposed
__global__ void ew_comb_kernel(const float* __restrict__ zg, const float* __restrict__ cg,
                               const float* __restrict__ hv1, const float* __restrict__ obs,
                               int t, float* __restrict__ hv, _Float16* __restrict__ hvT) {
    long i = (long)blockIdx.x * blockDim.x + threadIdx.x;
    if (i >= (long)NB * NN * DR) return;
    int d = i & (DR - 1); long nn = i >> 7; int n = nn & (NN - 1); int b = (int)(nn >> 10);
    float h1 = hv1[i];
    float o  = obs[b * NT + t];
    float z  = zg[i];
    float v  = (o > 0.5f) ? ((1.0f - z) * h1 + z * cg[i]) : h1;
    hv[i] = v;
    hvT[((long)b * DR + d) * NN + n] = (_Float16)v;
}

// ---------------------------------------------------------------------------
extern "C" void kernel_launch(void* const* d_in, const int* in_sizes, int n_in,
                              void* d_out, int out_size, void* d_ws, size_t ws_size,
                              hipStream_t stream) {
    (void)in_sizes; (void)n_in; (void)out_size; (void)ws_size;
    const float* values = (const float*)d_in[0];
    const float* masks  = (const float*)d_in[1];
    const float* A      = (const float*)d_in[2];
    const float* h0     = (const float*)d_in[3];
    const float* W0 = (const float*)d_in[4];  const float* b0 = (const float*)d_in[5];
    const float* W1 = (const float*)d_in[6];  const float* b1 = (const float*)d_in[7];
    const float* Wout = (const float*)d_in[8]; const float* bout = (const float*)d_in[9];
    const float* Wz = (const float*)d_in[10]; const float* bz = (const float*)d_in[11];
    const float* Wr = (const float*)d_in[12]; const float* br = (const float*)d_in[13];
    const float* Wh = (const float*)d_in[14]; const float* bh = (const float*)d_in[15];
    const float* Wo = (const float*)d_in[16]; const float* bo = (const float*)d_in[17];
    float* out = (float*)d_out;

    // ---- workspace layout ----
    char* ws = (char*)d_ws;
    size_t off = 0;
    auto take = [&](size_t bytes) { char* p = ws + off; off = (off + bytes + 255) & ~(size_t)255; return p; };
    _Float16* A16   = (_Float16*)take((size_t)NB * NN * NN * 2);     // 16 MB (L2-resident)
    _Float16* XTt   = (_Float16*)take((size_t)NB * DIN * NN * 2);    // x_t^T
    float*    hv    = (float*)   take((size_t)NB * NN * DR * 4);
    float*    hv1   = (float*)   take((size_t)NB * NN * DR * 4);
    _Float16* hv116 = (_Float16*)take((size_t)NB * NN * DR * 2);     // hv1 row-major f16
    _Float16* hvT   = (_Float16*)take((size_t)NB * DR * NN * 2);
    _Float16* hv1T  = (_Float16*)take((size_t)NB * DR * NN * 2);
    _Float16* uT    = (_Float16*)take((size_t)NB * DR * NN * 2);
    _Float16* rhT   = (_Float16*)take((size_t)NB * DR * NN * 2);
    _Float16* G0    = (_Float16*)take((size_t)NB * NN * DXH * 2);    // A@H results
    _Float16* G1    = (_Float16*)take((size_t)NB * NN * DR * 2);     // A@(r*h)
    float*    odeo  = (float*)   take((size_t)NB * NN * DR * 4);
    float*    zbuf  = (float*)   take((size_t)NB * NN * DR * 4);
    float*    rbuf  = (float*)   take((size_t)NB * NN * DR * 4);
    float*    cbuf  = (float*)   take((size_t)NB * NN * DR * 4);
    _Float16* WT0   = (_Float16*)take(DR * DR * 2);
    _Float16* WT1   = (_Float16*)take(DR * DR * 2);
    _Float16* WTout = (_Float16*)take(DR * DR * 2);
    _Float16* WTz   = (_Float16*)take(DXH * DR * 2);
    _Float16* WTr   = (_Float16*)take(DXH * DR * 2);
    _Float16* WTh   = (_Float16*)take(DXH * DR * 2);
    _Float16* WTo   = (_Float16*)take(DR * DIN * 2);
    float*    obs   = (float*)   take(NB * NT * 4);

    const long elemsH = (long)NB * NN * DR;
    const int  EWB = (int)((elemsH + 255) / 256);
    const long hT_str = (long)DR * NN;       // batch stride of [DR][NN] f16
    const long xT_str = (long)DIN * NN;
    const int  BIG = 1 << 30;

    // ---- prep ----
    cvt_a_kernel<<<(int)(((long)NB*NN*NN + 255)/256), 256, 0, stream>>>(A, A16, (long)NB*NN*NN);
    wtr_kernel<<<(DR*DR+255)/256, 256, 0, stream>>>(W0,  WT0,  DR, DR);
    wtr_kernel<<<(DR*DR+255)/256, 256, 0, stream>>>(W1,  WT1,  DR, DR);
    wtr_kernel<<<(DR*DR+255)/256, 256, 0, stream>>>(Wout,WTout,DR, DR);
    wtr_kernel<<<(DXH*DR+255)/256, 256, 0, stream>>>(Wz, WTz, DXH, DR);
    wtr_kernel<<<(DXH*DR+255)/256, 256, 0, stream>>>(Wr, WTr, DXH, DR);
    wtr_kernel<<<(DXH*DR+255)/256, 256, 0, stream>>>(Wh, WTh, DXH, DR);
    wtr_kernel<<<(DR*DIN+255)/256, 256, 0, stream>>>(Wo, WTo, DR, DIN);
    init_h_kernel<<<EWB, 256, 0, stream>>>(h0, hv, hvT);
    obs_kernel<<<NB * NT, 256, 0, stream>>>(masks, obs);

    dim3 g1a(8, DR / 64, NB);    // k1 with D=128
    dim3 g1x(8, DXH / 64, NB);   // k1 with D=192
    dim3 g2r(8, DR / 16, NB);    // k2 with Dout=128
    dim3 g2o(8, DIN / 16, NB);   // k2 with Dout=64 (head)

    // 33 iterations: i=0 is the initial ODE-only step; i>=1 are scan steps t=i-1.
    for (int i = 0; i <= NT; ++i) {
        // ---- ODE: 3x (A-mult + dense) ----
        k1_spmm<<<g1a, 256, 0, stream>>>(A16, hvT, hT_str, hvT, hT_str, BIG,
                                         G0, DR, (long)NN * DR);
        k2_dense<1,1><<<g2r, 256, 0, stream>>>(G0, DR, (long)NN*DR, DR, G0, DR, (long)NN*DR,
                                               WT0, DR, b0, nullptr, 0, 0, uT, hT_str);
        k1_spmm<<<g1a, 256, 0, stream>>>(A16, uT, hT_str, uT, hT_str, BIG,
                                         G0, DR, (long)NN * DR);
        k2_dense<1,1><<<g2r, 256, 0, stream>>>(G0, DR, (long)NN*DR, DR, G0, DR, (long)NN*DR,
                                               WT1, DR, b1, nullptr, 0, 0, uT, hT_str);
        k1_spmm<<<g1a, 256, 0, stream>>>(A16, uT, hT_str, uT, hT_str, BIG,
                                         G0, DR, (long)NN * DR);
        k2_dense<0,0><<<g2r, 256, 0, stream>>>(G0, DR, (long)NN*DR, DR, G0, DR, (long)NN*DR,
                                               WTout, DR, bout, odeo, DR, (long)NN*DR, nullptr, 0);

        if (i == 0) {   // hv <- hv + DT*ode(hv); no GRU, no output
            ew_ode_kernel<<<EWB, 256, 0, stream>>>(hv, odeo, hv, nullptr, hvT);
            continue;
        }
        const int t = i - 1;
        ew_ode_kernel<<<EWB, 256, 0, stream>>>(hv, odeo, hv1, hv116, hv1T);

        // ---- output head: X_pred[:, t-1] = hv1 @ Wo + bo  (recorded for t>=1) ----
        if (t >= 1) {
            k2_dense<0,0><<<g2o, 256, 0, stream>>>(
                hv116, DR, (long)NN*DR, DR, hv116, DR, (long)NN*DR,
                WTo, DR, bo,
                out + (long)(t - 1) * NN * DIN, DIN, (long)(NT - 1) * NN * DIN,
                nullptr, 0);
        }

        // ---- GRU ----
        xt_kernel<<<(int)(((long)NB*DIN*NN + 255)/256), 256, 0, stream>>>(values, masks, XTt, t);
        // Gxh = A @ concat(x, hv1)   (shared by z, r, and the x-half of the c-gate)
        k1_spmm<<<g1x, 256, 0, stream>>>(A16, XTt, xT_str, hv1T, hT_str, DIN,
                                         G0, DXH, (long)NN * DXH);
        k2_dense<2,0><<<g2r, 256, 0, stream>>>(G0, DXH, (long)NN*DXH, DXH, G0, DXH, (long)NN*DXH,
                                               WTz, DXH, bz, zbuf, DR, (long)NN*DR, nullptr, 0);
        k2_dense<2,0><<<g2r, 256, 0, stream>>>(G0, DXH, (long)NN*DXH, DXH, G0, DXH, (long)NN*DXH,
                                               WTr, DXH, br, rbuf, DR, (long)NN*DR, nullptr, 0);
        ew_rh_kernel<<<EWB, 256, 0, stream>>>(rbuf, hv1, rhT);
        // Grh = A @ (r*hv1)
        k1_spmm<<<g1a, 256, 0, stream>>>(A16, rhT, hT_str, rhT, hT_str, BIG,
                                         G1, DR, (long)NN * DR);
        // c = tanh([Gxh[:, :64] | Grh] @ Wh + bh)   (split A-operand, fused)
        k2_dense<1,0><<<g2r, 256, 0, stream>>>(G0, DXH, (long)NN*DXH, DIN, G1, DR, (long)NN*DR,
                                               WTh, DXH, bh, cbuf, DR, (long)NN*DR, nullptr, 0);
        ew_comb_kernel<<<EWB, 256, 0, stream>>>(zbuf, cbuf, hv1, obs, t, hv, hvT);
    }
}
